// AttentionBlock_75505525063697
// MI455X (gfx1250) — matrix operand order repaired and checked by reading, not compiled, not run on hardware
//
#include <hip/hip_runtime.h>

#define B_   8
#define C_   128
#define HW_  4096
#define G_   32
#define EPS_ 1e-5f
// 128^-0.5 * log2(e): attention logits produced directly in exp2 domain
#define QSCALE_ 0.12751664804056682f
#define SCALE_  0.08838834764831845f

typedef __attribute__((ext_vector_type(16))) __bf16       v16bf;
typedef __attribute__((ext_vector_type(8)))  float        v8f;
typedef __attribute__((ext_vector_type(4)))  unsigned int u32x4;

union BF16x16 { v16bf v; __bf16 e[16]; u32x4 q[2]; };
union BF16x8  { u32x4 q; unsigned short e[8]; __bf16 b[8]; };

__device__ __forceinline__ unsigned short f2bf_u(float f) {
  unsigned u = __builtin_bit_cast(unsigned, f);
  unsigned r = u + 0x7FFFu + ((u >> 16) & 1u);     // round-to-nearest-even
  return (unsigned short)(r >> 16);
}
__device__ __forceinline__ v8f wmma_bf16(v16bf a, v16bf b, v8f c) {
  // D = A(16x32 bf16) x B(32x16 bf16) + C(16x16 f32)
  return __builtin_amdgcn_wmma_f32_16x16x32_bf16(false, a, false, b, (short)0, c, false, false);
}

// ---------------- GroupNorm statistics: one block per (b, group) ----------------
__global__ __launch_bounds__(256) void gn_stats(const float* __restrict__ x,
                                                float* __restrict__ stats) {
  const int bg = blockIdx.x;                 // 0..255
  const float* p = x + (size_t)bg * 16384;   // group = 4 contiguous channels * 4096
  float s = 0.f, ss = 0.f;
  for (int i = threadIdx.x; i < 16384; i += 256) { float v = p[i]; s += v; ss += v * v; }
  for (int off = 16; off; off >>= 1) { s += __shfl_xor(s, off, 32); ss += __shfl_xor(ss, off, 32); }
  __shared__ float sb[8], ssb[8];
  const int w = threadIdx.x >> 5;
  if ((threadIdx.x & 31) == 0) { sb[w] = s; ssb[w] = ss; }
  __syncthreads();
  if (threadIdx.x == 0) {
    float S = 0.f, SS = 0.f;
    for (int i = 0; i < 8; i++) { S += sb[i]; SS += ssb[i]; }
    float mean = S * (1.f / 16384.f);
    float var  = SS * (1.f / 16384.f) - mean * mean;
    stats[2 * bg]     = mean;
    stats[2 * bg + 1] = rsqrtf(var + EPS_);
  }
}

// ---------------- Convert 1x1-conv weights to bf16 ----------------
__global__ __launch_bounds__(256) void cvt_w(const float* __restrict__ wq,
                                             const float* __restrict__ wo,
                                             unsigned short* __restrict__ wqb,
                                             unsigned short* __restrict__ wob) {
  const int i = blockIdx.x * 256 + threadIdx.x;
  if (i < 384 * 128) wqb[i] = f2bf_u(wq[i]);
  if (i < 128 * 128) wob[i] = f2bf_u(wo[i]);
}

// ---------------- Normalize + transpose to pixel-major bf16 [b][p][c] ----------------
__global__ __launch_bounds__(256) void gn_norm_t(const float* __restrict__ x,
                                                 const float* __restrict__ stats,
                                                 const float* __restrict__ gamma,
                                                 const float* __restrict__ beta,
                                                 unsigned short* __restrict__ xn) {
  const int b  = blockIdx.z;
  const int c0 = blockIdx.y * 32;
  const int p0 = blockIdx.x * 32;
  __shared__ float tile[32][33];
  const int tp  = threadIdx.x & 31;
  const int tc8 = threadIdx.x >> 5;
#pragma unroll
  for (int i = 0; i < 4; i++) {
    const int c = c0 + tc8 + i * 8;
    const int sidx = 2 * (b * G_ + (c >> 2));
    const float mean = stats[sidx];
    const float rstd = stats[sidx + 1];
    const float v = x[((size_t)(b * C_ + c)) * HW_ + p0 + tp];
    tile[tc8 + i * 8][tp] = (v - mean) * rstd * gamma[c] + beta[c];
  }
  __syncthreads();
  const int cc  = threadIdx.x & 31;
  const int pp8 = threadIdx.x >> 5;
#pragma unroll
  for (int i = 0; i < 4; i++) {
    const int p = p0 + pp8 + i * 8;
    xn[((size_t)(b * HW_ + p)) * C_ + c0 + cc] = f2bf_u(tile[cc][pp8 + i * 8]);
  }
}

// ---------------- QKV projection GEMM (bf16 WMMA) ----------------
// qkv[o][p] = sum_c W[o][c] * xn[p][c] ; Q,K stored pixel-major, V channel-major.
__global__ __launch_bounds__(128) void qkv_gemm(const unsigned short* __restrict__ wbf,
                                                const float* __restrict__ bqkv,
                                                const unsigned short* __restrict__ xn,
                                                unsigned short* __restrict__ q,
                                                unsigned short* __restrict__ k,
                                                unsigned short* __restrict__ v) {
  const int wave = blockIdx.x * 4 + (threadIdx.x >> 5);   // 49152 waves
  const int lane = threadIdx.x & 31;
  const int n  = lane & 15;
  const int hi = lane >> 4;
  const int mb = hi * 8;
  const int pt   = wave & 255;
  const int rest = wave >> 8;           // 0..191
  const int ot   = rest % 24;
  const int b    = rest / 24;

  const unsigned short* wrow = wbf + (size_t)(ot * 16 + n) * C_ + (hi ? 8 : 0);
  const unsigned short* xrow = xn + ((size_t)(b * HW_ + pt * 16 + n)) * C_ + hi * 8;

  v8f acc = (v8f)0.f;
#pragma unroll
  for (int ccn = 0; ccn < 4; ccn++) {
    BF16x16 a, bb;
    a.q[0]  = *(const u32x4*)(wrow + ccn * 32);        // K {0..7} or {8..15}
    a.q[1]  = *(const u32x4*)(wrow + ccn * 32 + 16);   // K {16..23} or {24..31}
    bb.q[0] = *(const u32x4*)(xrow + ccn * 32);        // K {0..7}/{16..23} or {8..15}/{24..31}
    bb.q[1] = *(const u32x4*)(xrow + ccn * 32 + 16);
    acc = wmma_bf16(a.v, bb.v, acc);
  }

  const int obase = ot * 16 + mb;
  const int p = pt * 16 + n;
  float vals[8];
#pragma unroll
  for (int r = 0; r < 8; r++) vals[r] = acc[r] + bqkv[obase + r];

  if (ot < 8) {                                  // Q (1/sqrt(C) * log2e folded here)
    BF16x8 pk;
#pragma unroll
    for (int r = 0; r < 8; r++) pk.e[r] = f2bf_u(vals[r] * QSCALE_);
    *(u32x4*)(q + ((size_t)(b * HW_ + p)) * C_ + obase) = pk.q;
  } else if (ot < 16) {                          // K
    BF16x8 pk;
#pragma unroll
    for (int r = 0; r < 8; r++) pk.e[r] = f2bf_u(vals[r]);
    *(u32x4*)(k + ((size_t)(b * HW_ + p)) * C_ + (obase - 128)) = pk.q;
  } else {                                       // V channel-major
#pragma unroll
    for (int r = 0; r < 8; r++)
      v[((size_t)(b * C_ + (obase - 256) + r)) * HW_ + p] = f2bf_u(vals[r]);
  }
}

// ---------------- Flash attention: one 16-query tile per wave ----------------
// Logits are in exp2 domain (log2e folded into Q); softmax uses v_exp_f32 directly.
__global__ __launch_bounds__(128) void attn_kernel(const unsigned short* __restrict__ qb,
                                                   const unsigned short* __restrict__ kbuf,
                                                   const unsigned short* __restrict__ vbuf,
                                                   unsigned short* __restrict__ ob) {
  const int wave = blockIdx.x * 4 + (threadIdx.x >> 5);   // 2048 waves
  const int lane = threadIdx.x & 31;
  const int b  = wave >> 8;
  const int qt = wave & 255;
  const int n  = lane & 15;
  const int hi = lane >> 4;
  const int mb = hi * 8;

  // Persistent Q B-operand (32chan x 16query tiles), 4 chunks over C=128.
  BF16x16 qB[4];
  const unsigned short* qrow = qb + ((size_t)(b * HW_ + qt * 16 + n)) * C_ + hi * 8;
#pragma unroll
  for (int ccn = 0; ccn < 4; ccn++) {
    qB[ccn].q[0] = *(const u32x4*)(qrow + ccn * 32);
    qB[ccn].q[1] = *(const u32x4*)(qrow + ccn * 32 + 16);
  }

  v8f o[8];
#pragma unroll
  for (int c = 0; c < 8; c++) o[c] = (v8f)0.f;
  float m_run = -1e30f, l_run = 0.f;

  const unsigned short* krow0 = kbuf + ((size_t)b * HW_) * C_;
  const unsigned short* vbase = vbuf + ((size_t)b * C_) * HW_;

  for (int kb0 = 0; kb0 < HW_; kb0 += 32) {
    // S^T = K * Q^T : D-layout(lane=query, VGPR=key) == A-layout for the PV WMMA
    v8f s0 = (v8f)0.f, s1 = (v8f)0.f;
    const unsigned short* k0 = krow0 + (size_t)(kb0 + n) * C_ + (hi ? 8 : 0);
    const unsigned short* k1 = k0 + 16 * C_;
#pragma unroll
    for (int ccn = 0; ccn < 4; ccn++) {
      BF16x16 a0, a1;
      a0.q[0] = *(const u32x4*)(k0 + ccn * 32);
      a0.q[1] = *(const u32x4*)(k0 + ccn * 32 + 16);
      a1.q[0] = *(const u32x4*)(k1 + ccn * 32);
      a1.q[1] = *(const u32x4*)(k1 + ccn * 32 + 16);
      s0 = wmma_bf16(a0.v, qB[ccn].v, s0);
      s1 = wmma_bf16(a1.v, qB[ccn].v, s1);
    }

    // Online softmax over this 32-key slab.
    float tmax = s0[0];
#pragma unroll
    for (int r = 1; r < 8; r++) tmax = fmaxf(tmax, s0[r]);
#pragma unroll
    for (int r = 0; r < 8; r++) tmax = fmaxf(tmax, s1[r]);
    tmax = fmaxf(tmax, __shfl_xor(tmax, 16, 32));
    const float m_new = fmaxf(m_run, tmax);

    // Lazy rescale: wave-uniform branch (EXEC stays full); taken only while the
    // running max is still growing, so steady-state iterations skip 64 muls +
    // 8 bpermutes entirely. alpha == 1 exactly for queries whose max didn't move.
    if (__ballot(tmax > m_run)) {
      const float alpha = __builtin_amdgcn_exp2f(m_run - m_new);
      float alpha_o[8];
#pragma unroll
      for (int r = 0; r < 8; r++)
        alpha_o[r] = __shfl(alpha, (lane < 16) ? r : (r + 8), 32);
#pragma unroll
      for (int c = 0; c < 8; c++)
#pragma unroll
        for (int r = 0; r < 8; r++) o[c][r] *= alpha_o[r];
      l_run *= alpha;
    }
    m_run = m_new;

    BF16x16 pA;
    float lsum = 0.f;
#pragma unroll
    for (int r = 0; r < 8; r++) {
      float e0 = __builtin_amdgcn_exp2f(s0[r] - m_new); lsum += e0; pA.e[r]     = (__bf16)e0;
      float e1 = __builtin_amdgcn_exp2f(s1[r] - m_new); lsum += e1; pA.e[8 + r] = (__bf16)e1;
    }
    lsum += __shfl_xor(lsum, 16, 32);
    l_run += lsum;

#pragma unroll
    for (int c = 0; c < 8; c++) {
      BF16x16 vB;
      const unsigned short* vrow = vbase + (size_t)(c * 16 + n) * HW_ + kb0 + hi * 8;
      vB.q[0] = *(const u32x4*)(vrow);
      vB.q[1] = *(const u32x4*)(vrow + 16);
      o[c] = wmma_bf16(pA.v, vB.v, o[c]);
    }
  }

  const float rinv = 1.f / l_run;
  float rinv_o[8];
#pragma unroll
  for (int r = 0; r < 8; r++)
    rinv_o[r] = __shfl(rinv, (lane < 16) ? r : (r + 8), 32);

  unsigned short* orow = ob + ((size_t)(b * HW_ + qt * 16)) * C_;
#pragma unroll
  for (int c = 0; c < 8; c++) {
#pragma unroll
    for (int r = 0; r < 8; r++) {
      const int m = mb + r;
      const __bf16 bv = (__bf16)(o[c][r] * rinv_o[r]);
      orow[(size_t)m * C_ + c * 16 + n] = __builtin_bit_cast(unsigned short, bv);
    }
  }
}

// ---------------- Output projection + bias + residual ----------------
__global__ __launch_bounds__(128) void out_gemm(const unsigned short* __restrict__ wobf,
                                                const float* __restrict__ bout,
                                                const unsigned short* __restrict__ attn,
                                                const float* __restrict__ x,
                                                float* __restrict__ out) {
  const int wave = blockIdx.x * 4 + (threadIdx.x >> 5);   // 16384 waves
  const int lane = threadIdx.x & 31;
  const int n  = lane & 15;
  const int hi = lane >> 4;
  const int mb = hi * 8;
  const int pt   = wave & 255;
  const int rest = wave >> 8;     // 0..63
  const int ot   = rest & 7;
  const int b    = rest >> 3;

  const unsigned short* wrow = wobf + (size_t)(ot * 16 + n) * C_ + (hi ? 8 : 0);
  const unsigned short* arow = attn + ((size_t)(b * HW_ + pt * 16 + n)) * C_ + hi * 8;

  v8f acc = (v8f)0.f;
#pragma unroll
  for (int ccn = 0; ccn < 4; ccn++) {
    BF16x16 a, bb;
    a.q[0]  = *(const u32x4*)(wrow + ccn * 32);
    a.q[1]  = *(const u32x4*)(wrow + ccn * 32 + 16);
    bb.q[0] = *(const u32x4*)(arow + ccn * 32);
    bb.q[1] = *(const u32x4*)(arow + ccn * 32 + 16);
    acc = wmma_bf16(a.v, bb.v, acc);
  }

  const int obase = ot * 16 + mb;
  const int p = pt * 16 + n;
#pragma unroll
  for (int r = 0; r < 8; r++) {
    const int oc = obase + r;
    const size_t idx = ((size_t)(b * C_ + oc)) * HW_ + p;
    out[idx] = acc[r] + bout[oc] + x[idx];
  }
}

extern "C" void kernel_launch(void* const* d_in, const int* in_sizes, int n_in,
                              void* d_out, int out_size, void* d_ws, size_t ws_size,
                              hipStream_t stream) {
  (void)in_sizes; (void)n_in; (void)out_size; (void)ws_size;
  const float* x     = (const float*)d_in[0];
  const float* gamma = (const float*)d_in[1];
  const float* beta  = (const float*)d_in[2];
  const float* w_qkv = (const float*)d_in[3];
  const float* b_qkv = (const float*)d_in[4];
  const float* w_out = (const float*)d_in[5];
  const float* b_out = (const float*)d_in[6];
  float* out = (float*)d_out;

  char* ws = (char*)d_ws;
  size_t off = 0;
  float* stats = (float*)(ws + off);              off += 2048;
  unsigned short* wqb = (unsigned short*)(ws + off); off += (size_t)384 * 128 * 2;
  unsigned short* wob = (unsigned short*)(ws + off); off += (size_t)128 * 128 * 2;
  off = (off + 255) & ~(size_t)255;
  const size_t TEN = (size_t)B_ * HW_ * C_ * 2;   // 8 MB per bf16 tensor
  unsigned short* xnb = (unsigned short*)(ws + off); off += TEN;
  unsigned short* qb  = (unsigned short*)(ws + off); off += TEN;
  unsigned short* kb  = (unsigned short*)(ws + off); off += TEN;
  unsigned short* vb  = (unsigned short*)(ws + off); off += TEN;
  unsigned short* ab  = (unsigned short*)(ws + off); off += TEN;

  gn_stats<<<256, 256, 0, stream>>>(x, stats);
  cvt_w<<<192, 256, 0, stream>>>(w_qkv, w_out, wqb, wob);
  gn_norm_t<<<dim3(128, 4, 8), 256, 0, stream>>>(x, stats, gamma, beta, xnb);
  qkv_gemm<<<12288, 128, 0, stream>>>(wqb, b_qkv, xnb, qb, kb, vb);
  attn_kernel<<<512, 128, 0, stream>>>(qb, kb, vb, ab);
  out_gemm<<<4096, 128, 0, stream>>>(wob, b_out, ab, x, out);
}
